// WritheMessage_37632503448184
// MI455X (gfx1250) — matrix operand order repaired and checked by reading, not compiled, not run on hardware
//
#include <hip/hip_runtime.h>
#include <hip/hip_fp16.h>
#include <math.h>

typedef __attribute__((ext_vector_type(16))) _Float16 v16h;
typedef __attribute__((ext_vector_type(8)))  _Float16 v8h;
typedef __attribute__((ext_vector_type(8)))  float    v8f;

#define N_ATOMS 256
#define N_FEAT  128
#define BATCH   8
#define BINS    300
#define KPAD    320               // 300 padded to multiple of 32 (WMMA K)
#define NSEG    32131
#define TOT     (BATCH * NSEG)    // 257048 segment-rows
#define NNODES  (BATCH * N_ATOMS) // 2048
#define STEPF   (2.0f / (BINS - 1))
#define MROWS   32                // segment-rows per block (two 16-row M tiles)

// workspace layout (bytes), all 256-aligned
#define OFF_WR    0            // TOT floats      (1,028,192 B)
#define OFF_DENOM 1048576      // NNODES floats   (8,192 B)
#define OFF_SEG   1056768      // NSEG int2       (257,048 B)
#define OFF_BASIS 1318912      // 128*KPAD f16    (81,920 B)

struct F3 { float x, y, z; };
__device__ __forceinline__ F3 f3sub(F3 a, F3 b) { return {a.x-b.x, a.y-b.y, a.z-b.z}; }
__device__ __forceinline__ float f3dot(F3 a, F3 b) { return a.x*b.x + a.y*b.y + a.z*b.z; }
__device__ __forceinline__ F3 f3cross(F3 a, F3 b) {
    return {a.y*b.z - a.z*b.y, a.z*b.x - a.x*b.z, a.x*b.y - a.y*b.x};
}
__device__ __forceinline__ F3 f3norm(F3 a) {
    float r = rsqrtf(f3dot(a, a));
    return {a.x*r, a.y*r, a.z*r};
}
__device__ __forceinline__ F3 ldc(const float* c, int i) {
    return {c[3*i], c[3*i+1], c[3*i+2]};
}
__device__ __forceinline__ int segC(int i) { return (i * (507 - i)) >> 1; } // pairs before first-index i

// ---------------------------------------------------------------------------
// Prep: out <- node_features, denom <- 0, basis -> f16 transposed [n][k],
// triangular segment table (i,j) from linear index.
// ---------------------------------------------------------------------------
__global__ void k_prep(const float* __restrict__ nf, float* __restrict__ out,
                       float* __restrict__ denom, const float* __restrict__ basis,
                       _Float16* __restrict__ basisT, int2* __restrict__ seg) {
    int idx = blockIdx.x * 256 + threadIdx.x;
    if (idx < NNODES * N_FEAT) out[idx] = nf[idx];
    if (idx < NNODES) denom[idx] = 0.0f;
    if (idx < N_FEAT * KPAD) {
        int n = idx / KPAD, k = idx - n * KPAD;
        basisT[idx] = (k < BINS) ? (_Float16)basis[k * N_FEAT + n] : (_Float16)0.0f;
    }
    if (idx < NSEG) {
        // invert t = C(i) + (j - i - 2): i = floor((507 - sqrt(507^2 - 8t))/2) + fixup
        int i = (int)((507.0 - sqrt(257049.0 - 8.0 * (double)idx)) * 0.5);
        if (i < 0) i = 0; if (i > 252) i = 252;
        while (i > 0 && segC(i) > idx) --i;
        while (i < 252 && segC(i + 1) <= idx) ++i;
        int j = i + 2 + (idx - segC(i));
        seg[idx] = make_int2(i, j);
    }
}

// ---------------------------------------------------------------------------
// Writhe per (batch, segment-pair) + attention denominator accumulation.
// ---------------------------------------------------------------------------
__global__ void k_writhe(const float* __restrict__ coords, const int2* __restrict__ seg,
                         float* __restrict__ wr, float* __restrict__ denom) {
    int t = blockIdx.x * 256 + threadIdx.x;
    if (t >= TOT) return;
    int b = t / NSEG, s = t - b * NSEG;
    int2 ij = seg[s];
    int base = b * N_ATOMS;
    F3 p0 = ldc(coords, base + ij.x);
    F3 p1 = ldc(coords, base + ij.x + 1);
    F3 p2 = ldc(coords, base + ij.y);
    F3 p3 = ldc(coords, base + ij.y + 1);

    F3 dx0 = f3norm(f3sub(p2, p0));
    F3 dx1 = f3norm(f3sub(p3, p0));
    F3 dx2 = f3norm(f3sub(p2, p1));
    F3 dx3 = f3norm(f3sub(p3, p1));

    F3 cr0 = f3norm(f3cross(dx0, dx1));
    F3 cr1 = f3norm(f3cross(dx1, dx3));
    F3 cr2 = f3norm(f3cross(dx3, dx2));
    F3 cr3 = f3norm(f3cross(dx2, dx0));

    float d0 = fminf(1.0f, fmaxf(-1.0f, f3dot(cr0, cr1)));
    float d1 = fminf(1.0f, fmaxf(-1.0f, f3dot(cr1, cr2)));
    float d2 = fminf(1.0f, fmaxf(-1.0f, f3dot(cr2, cr3)));
    float d3 = fminf(1.0f, fmaxf(-1.0f, f3dot(cr3, cr0)));
    float omega = asinf(d0) + asinf(d1) + asinf(d2) + asinf(d3);

    F3 sv = f3cross(f3sub(p3, p2), f3sub(p1, p0));
    float sd = f3dot(sv, dx0);
    float sgn = (sd > 0.0f) ? 1.0f : ((sd < 0.0f) ? -1.0f : 0.0f);
    wr[t] = omega * sgn * 0.15915494309189535f; // 1/(2*pi)

    // attention weights for the two edges of this segment pair (dst = j, j+1)
    F3 e1 = f3sub(p0, p2);
    F3 e2 = f3sub(p1, p3);
    atomicAdd(&denom[base + ij.y],     __expf(-f3dot(e1, e1)));
    atomicAdd(&denom[base + ij.y + 1], __expf(-f3dot(e2, e2)));
}

// ---------------------------------------------------------------------------
// Fused RBF-encode (LDS f16) -> WMMA GEMM vs basis -> attention scatter.
// Block: 128 threads (4 waves), 32 segment-rows (2 M-tiles), 128 feature cols.
// Each wave: 2 N-tiles x 2 M-tiles = 4 accumulators; B fragments reused
// across both M-tiles (2 B loads : 4 wmma per K-step).
// ---------------------------------------------------------------------------
__global__ __launch_bounds__(128)
void k_wmma(const float* __restrict__ coords, const _Float16* __restrict__ basisT,
            const int2* __restrict__ seg, const float* __restrict__ wr,
            const float* __restrict__ denom, float* __restrict__ out) {
    __shared__ __align__(32) _Float16 encs[MROWS * KPAD]; // 20 KB
    __shared__ float wrs[MROWS];
    __shared__ float attnA[MROWS], attnB[MROWS];
    __shared__ int   dstA[MROWS],  dstB[MROWS];

    int tid = threadIdx.x;
    int s0  = blockIdx.x * MROWS;

    if (tid < MROWS) {
        int row = s0 + tid;
        float wv = 0.0f, a1 = 0.0f, a2 = 0.0f;
        int   d1 = 0,    d2 = 0;
        if (row < TOT) {
            wv = wr[row];
            int b = row / NSEG, s = row - b * NSEG;
            int2 ij = seg[s];
            int base = b * N_ATOMS;
            F3 p0 = ldc(coords, base + ij.x);
            F3 p1 = ldc(coords, base + ij.x + 1);
            F3 p2 = ldc(coords, base + ij.y);
            F3 p3 = ldc(coords, base + ij.y + 1);
            F3 e1 = f3sub(p0, p2);
            F3 e2 = f3sub(p1, p3);
            d1 = base + ij.y;
            d2 = base + ij.y + 1;
            a1 = __expf(-f3dot(e1, e1)) / denom[d1];
            a2 = __expf(-f3dot(e2, e2)) / denom[d2];
        }
        wrs[tid] = wv; attnA[tid] = a1; attnB[tid] = a2; dstA[tid] = d1; dstB[tid] = d2;
    }
    __syncthreads();

    // Gaussian soft one-hot tile, f16 in LDS: enc[m][k], k >= BINS zero-padded
    const float inv112  = 1.0f / 1.12f;
    const float invstep = 1.0f / STEPF;
    for (int idx = tid; idx < MROWS * KPAD; idx += 128) {
        int m = idx / KPAD, k = idx - m * KPAD;
        float v = 0.0f;
        if (k < BINS) {
            float c = -1.0f + (float)k * STEPF;
            float x = (wrs[m] - c) * invstep;
            v = __expf(-x * x) * inv112;
        }
        encs[idx] = (_Float16)v;
    }
    __syncthreads();

    int wave = tid >> 5;
    int lane = tid & 31;
    int nl   = lane & 15;          // N within 16x16 tile / A-matrix row M
    int hi   = (lane >> 4) & 1;    // lane-group selector for K halves
    int n0a  = wave * 32;          // this wave's first 16-col tile
    int n0b  = n0a + 16;           // second 16-col tile

    v8f acc00 = {}; v8f acc01 = {}; // M-tile 0 (rows 0..15)  x N-tiles a/b
    v8f acc10 = {}; v8f acc11 = {}; // M-tile 1 (rows 16..31) x N-tiles a/b
    #pragma unroll
    for (int k0 = 0; k0 < KPAD; k0 += 32) {
        // A fragments (16x32 f16): lanes<16 hold K=[k0,k0+16),
        // lanes>=16 hold K=[k0+16,k0+32)
        const _Float16* pa0 = &encs[nl * KPAD + k0 + hi * 16];
        const _Float16* pa1 = &encs[(16 + nl) * KPAD + k0 + hi * 16];
        v8h a0lo = *(const v8h*)(pa0);
        v8h a0hi = *(const v8h*)(pa0 + 8);
        v8h a1lo = *(const v8h*)(pa1);
        v8h a1hi = *(const v8h*)(pa1 + 8);
        v16h a0, a1;
        #pragma unroll
        for (int h = 0; h < 8; ++h) {
            a0[h] = a0lo[h]; a0[h + 8] = a0hi[h];
            a1[h] = a1lo[h]; a1[h + 8] = a1hi[h];
        }
        // B fragments (32x16 f16): lane holds 16 consecutive K for its column,
        // lane-group selects K halves; basisT is [n][k] so one 32B load each
        v16h b0 = *(const v16h*)(basisT + (n0a + nl) * KPAD + k0 + hi * 16);
        v16h b1 = *(const v16h*)(basisT + (n0b + nl) * KPAD + k0 + hi * 16);
        acc00 = __builtin_amdgcn_wmma_f32_16x16x32_f16(false, a0, false, b0,
                                                       (short)0, acc00, false, false);
        acc01 = __builtin_amdgcn_wmma_f32_16x16x32_f16(false, a0, false, b1,
                                                       (short)0, acc01, false, false);
        acc10 = __builtin_amdgcn_wmma_f32_16x16x32_f16(false, a1, false, b0,
                                                       (short)0, acc10, false, false);
        acc11 = __builtin_amdgcn_wmma_f32_16x16x32_f16(false, a1, false, b1,
                                                       (short)0, acc11, false, false);
    }

    // C layout: lanes<16 -> M=r, lanes>=16 -> M=8+r; N = lane&15 (+tile offset).
    // Scatter emb rows into both edges' destination nodes with attention.
    int mb = hi * 8;
    #pragma unroll
    for (int r = 0; r < 8; ++r) {
        int m0 = mb + r;       // row in M-tile 0
        int m1 = 16 + m0;      // row in M-tile 1
        {
            float aA = attnA[m0], aB = attnB[m0];
            int   dA = dstA[m0],  dB = dstB[m0];
            float v0 = acc00[r],  v1 = acc01[r];
            atomicAdd(&out[dA * N_FEAT + n0a + nl], v0 * aA);
            atomicAdd(&out[dB * N_FEAT + n0a + nl], v0 * aB);
            atomicAdd(&out[dA * N_FEAT + n0b + nl], v1 * aA);
            atomicAdd(&out[dB * N_FEAT + n0b + nl], v1 * aB);
        }
        {
            float aA = attnA[m1], aB = attnB[m1];
            int   dA = dstA[m1],  dB = dstB[m1];
            float v0 = acc10[r],  v1 = acc11[r];
            atomicAdd(&out[dA * N_FEAT + n0a + nl], v0 * aA);
            atomicAdd(&out[dB * N_FEAT + n0a + nl], v0 * aB);
            atomicAdd(&out[dA * N_FEAT + n0b + nl], v1 * aA);
            atomicAdd(&out[dB * N_FEAT + n0b + nl], v1 * aB);
        }
    }
}

extern "C" void kernel_launch(void* const* d_in, const int* in_sizes, int n_in,
                              void* d_out, int out_size, void* d_ws, size_t ws_size,
                              hipStream_t stream) {
    const float* coords = (const float*)d_in[0]; // (2048, 3)
    const float* nf     = (const float*)d_in[1]; // (2048, 128)
    const float* basis  = (const float*)d_in[2]; // (300, 128)
    float* out = (float*)d_out;                  // (2048, 128)

    char* ws = (char*)d_ws;
    float*    wr     = (float*)(ws + OFF_WR);
    float*    denom  = (float*)(ws + OFF_DENOM);
    int2*     seg    = (int2*)(ws + OFF_SEG);
    _Float16* basisT = (_Float16*)(ws + OFF_BASIS);

    int prepN = NNODES * N_FEAT; // 262144, covers all prep ranges
    k_prep  <<<(prepN + 255) / 256, 256, 0, stream>>>(nf, out, denom, basis, basisT, seg);
    k_writhe<<<(TOT   + 255) / 256, 256, 0, stream>>>(coords, seg, wr, denom);
    k_wmma  <<<(TOT + MROWS - 1) / MROWS, 128, 0, stream>>>(coords, basisT, seg, wr, denom, out);
}
